// DLRM_Net_31825707664001
// MI455X (gfx1250) — matrix operand order repaired and checked by reading, not compiled, not run on hardware
//
#include <hip/hip_runtime.h>
#include <hip/hip_bf16.h>

// ---------------------------------------------------------------------------
// Fused DLRM forward for MI455X (gfx1250, wave32, WMMA 16x16x32 bf16).
// Kernel 1: one-shot weight repack fp32 -> bf16, zero-padded K, decoders
//           transposed to [N,K] (lives in d_ws, ~0.94 MB).
// Kernel 2: fused forward, 32 batch rows per block (two 16-row M-tiles):
//           each B fragment (2x global_load_b128) feeds TWO v_wmma ops.
//           LDS buffers overlaid by lifetime: 160 KB total.
// ---------------------------------------------------------------------------

typedef __attribute__((ext_vector_type(16))) __bf16         v16bf;
typedef __attribute__((ext_vector_type(8)))  float          v8f;
typedef __attribute__((ext_vector_type(4)))  unsigned int   u32x4;
typedef __attribute__((ext_vector_type(4)))  float          f32x4;

#define DEVINL static __device__ __forceinline__

DEVINL unsigned short f2bf(float f) {
  __bf16 b = (__bf16)f;
  return __builtin_bit_cast(unsigned short, b);
}
DEVINL float bf2f(unsigned short h) {
  __bf16 b = __builtin_bit_cast(__bf16, h);
  return (float)b;
}

struct AFrag { u32x4 lo, hi; };   // 32 bytes == v16bf

// ---- d_ws layout (units: ushort/bf16), all 16B aligned ----
constexpr int OFF_W0 = 0;                     // [512][32]  (bw0, K 13->32 pad)
constexpr int OFF_W1 = OFF_W0 + 512 * 32;     // [256][512] (bw1)
constexpr int OFF_W2 = OFF_W1 + 256 * 512;    // [128][256] (bw2)
constexpr int OFF_T0 = OFF_W2 + 128 * 256;    // [512][192] (tw0, K 164->192 pad)
constexpr int OFF_T1 = OFF_T0 + 512 * 192;    // [256][512] (tw1)
constexpr int OFF_BD = OFF_T1 + 256 * 512;    // [4][128][32]  (big_dec^T)
constexpr int OFF_SD = OFF_BD + 4 * 128 * 32; // [4][128][128] (small_dec^T)

// ---------------------------------------------------------------------------
// Kernel 1: repack weights to bf16 [N, Kpad] row-major.
// ---------------------------------------------------------------------------
__global__ void repack_weights(const float* __restrict__ bw0,
                               const float* __restrict__ bw1,
                               const float* __restrict__ bw2,
                               const float* __restrict__ tw0,
                               const float* __restrict__ tw1,
                               const float* __restrict__ big_dec,
                               const float* __restrict__ small_dec,
                               unsigned short* __restrict__ ws)
{
  const int gid = blockIdx.x * blockDim.x + threadIdx.x;
  const int gs  = gridDim.x * blockDim.x;
  for (int i = gid; i < 512 * 32; i += gs) {            // bw0 pad 13->32
    int n = i >> 5, k = i & 31;
    ws[OFF_W0 + i] = f2bf(k < 13 ? bw0[n * 13 + k] : 0.f);
  }
  for (int i = gid; i < 256 * 512; i += gs)             // bw1
    ws[OFF_W1 + i] = f2bf(bw1[i]);
  for (int i = gid; i < 128 * 256; i += gs)             // bw2
    ws[OFF_W2 + i] = f2bf(bw2[i]);
  for (int i = gid; i < 512 * 192; i += gs) {           // tw0 pad 164->192
    int n = i / 192, k = i % 192;
    ws[OFF_T0 + i] = f2bf(k < 164 ? tw0[n * 164 + k] : 0.f);
  }
  for (int i = gid; i < 256 * 512; i += gs)             // tw1
    ws[OFF_T1 + i] = f2bf(tw1[i]);
  for (int i = gid; i < 4 * 128 * 32; i += gs) {        // big_dec [32,128]->[128,32]
    int t = i >> 12, r = i & 4095, n = r >> 5, k = r & 31;
    ws[OFF_BD + i] = f2bf(big_dec[t * 4096 + k * 128 + n]);
  }
  for (int i = gid; i < 4 * 128 * 128; i += gs) {       // small_dec [128,128]^T
    int t = i >> 14, r = i & 16383, n = r >> 7, k = r & 127;
    ws[OFF_SD + i] = f2bf(small_dec[t * 16384 + k * 128 + n]);
  }
}

// ---------------------------------------------------------------------------
// One 16(N) x 32(M) slab: two 16x16 D tiles sharing each B fragment.
// A: bf16 row-major [32, Kpad] in LDS. W: bf16 [N, Kpad] row-major global.
// Fragment layouts per CDNA5 ISA 7.12.2 (wave32):
//   A: lane l -> row m=l&15; holds K = h*8+e (e<8) and 16+h*8+e, h=l>>4
//   B: lane l -> col n=l&15; holds K = h*8+e (e<8) and 16+h*8+e (same as A)
//   D: VGPR v -> row m = v + h*8, col n = l&15
// ---------------------------------------------------------------------------
template<bool BIAS, bool RELU>
DEVINL void wmma_tile2(const unsigned short* __restrict__ A, int lda,
                       const unsigned short* __restrict__ W, int ldw,
                       int Kpad, int n0,
                       const float* __restrict__ bias,
                       unsigned short* __restrict__ out, int ldo, int lane)
{
  v8f c0 = {0.f, 0.f, 0.f, 0.f, 0.f, 0.f, 0.f, 0.f};
  v8f c1 = c0;
  const int m  = lane & 15;
  const int h  = lane >> 4;      // lane half: 0 or 1
  const int n  = lane & 15;
  const unsigned short* wrow  = W + (size_t)(n0 + n) * ldw + h * 8;
  const unsigned short* arow0 = A + (size_t)m * lda;          // rows 0..15
  const unsigned short* arow1 = A + (size_t)(16 + m) * lda;   // rows 16..31

  for (int k0 = 0; k0 < Kpad; k0 += 32) {
    AFrag bfr, a0, a1;
    bfr.lo = *(const u32x4*)(wrow + k0);
    bfr.hi = *(const u32x4*)(wrow + k0 + 16);
    a0.lo  = *(const u32x4*)(arow0 + k0 + h * 8);
    a0.hi  = *(const u32x4*)(arow0 + k0 + 16 + h * 8);
    a1.lo  = *(const u32x4*)(arow1 + k0 + h * 8);
    a1.hi  = *(const u32x4*)(arow1 + k0 + 16 + h * 8);
    if (k0 + 32 < Kpad)
      __builtin_prefetch(wrow + k0 + 32, 0, 3);   // global_prefetch_b8
    v16bf b  = __builtin_bit_cast(v16bf, bfr);
    v16bf av0 = __builtin_bit_cast(v16bf, a0);
    v16bf av1 = __builtin_bit_cast(v16bf, a1);
    c0 = __builtin_amdgcn_wmma_f32_16x16x32_bf16(
             false, av0, false, b, (short)0, c0, false, false);
    c1 = __builtin_amdgcn_wmma_f32_16x16x32_bf16(
             false, av1, false, b, (short)0, c1, false, false);
  }

  float bv = 0.f;
  if constexpr (BIAS) bv = bias[n0 + n];
#pragma unroll
  for (int v = 0; v < 8; ++v) {
    int   mo = v + h * 8;
    float r0 = c0[v] + bv, r1 = c1[v] + bv;
    if constexpr (RELU) {
      r0 = __builtin_fmaxf(r0, 0.f);
      r1 = __builtin_fmaxf(r1, 0.f);
    }
    out[(size_t)mo * ldo + n0 + n]        = f2bf(r0);
    out[(size_t)(16 + mo) * ldo + n0 + n] = f2bf(r1);
  }
}

// Dot-interaction for one batch row on the WMMA pipe:
// S = Zm * Zm^T with Zm = [16(features, 9 valid) x 128] bf16 in LDS.
// Lower triangle (i>j, i<9) written to Rrow as bf16 in tril order.
DEVINL void interact_row(const unsigned short* __restrict__ Zm,
                         unsigned short* __restrict__ Rrow, int lane)
{
  v8f c = {0.f, 0.f, 0.f, 0.f, 0.f, 0.f, 0.f, 0.f};
  const int i = lane & 15;
  const int h = lane >> 4;
#pragma unroll
  for (int k0 = 0; k0 < 128; k0 += 32) {
    AFrag af, bfr;
    af.lo  = *(const u32x4*)(Zm + (size_t)i * 128 + k0 + h * 8);
    af.hi  = *(const u32x4*)(Zm + (size_t)i * 128 + k0 + 16 + h * 8);
    bfr.lo = *(const u32x4*)(Zm + (size_t)i * 128 + k0 + h * 8);
    bfr.hi = *(const u32x4*)(Zm + (size_t)i * 128 + k0 + 16 + h * 8);
    v16bf a = __builtin_bit_cast(v16bf, af);
    v16bf b = __builtin_bit_cast(v16bf, bfr);
    c = __builtin_amdgcn_wmma_f32_16x16x32_bf16(
            false, a, false, b, (short)0, c, false, false);
  }
  const int fj = lane & 15;
#pragma unroll
  for (int v = 0; v < 8; ++v) {
    int fi = v + h * 8;
    if (fi < 9 && fj < fi)
      Rrow[fi * (fi - 1) / 2 + fj] = f2bf(c[v]);
  }
}

// ---------------------------------------------------------------------------

__global__ __launch_bounds__(128, 1)
void dlrm_fused(const float* __restrict__ dense_x,
                const int*   __restrict__ idx_big,
                const int*   __restrict__ idx_small,
                const float* __restrict__ big_tables,
                const float* __restrict__ small_tables,
                const unsigned short* __restrict__ ws,   // packed bf16 weights
                const float* __restrict__ bb0, const float* __restrict__ bb1,
                const float* __restrict__ bb2, const float* __restrict__ tb0,
                const float* __restrict__ tb1,
                const float* __restrict__ tw2, const float* __restrict__ tb2,
                float* __restrict__ out)
{
  constexpr int B = 4096, D = 128, DCMP = 32, L = 4;
  constexpr int NBIG = 4, NSML = 4;
  constexpr int ZS = 16 * 128;        // padded feature block per row

  // LDS overlay (lifetimes are disjoint; barriers separate phases):
  //   [0      , 16384) ACT0  -- also R (first 6144) and final T1 output
  //   [16384  , 81920) Z     -- first 16384 also ACT1 (L0 out / L1 in, T0 out / T1 in)
  __shared__ __align__(16) unsigned short lds[81920];    // 160 KB
  unsigned short* const sACT0 = lds;
  unsigned short* const sACT1 = lds + 16384;
  unsigned short* const sZ    = lds + 16384;
  unsigned short* const sR    = lds;

  const int tid  = threadIdx.x;
  const int lane = tid & 31;
  const int wave = tid >> 5;
  const int r0   = blockIdx.x * 32;

  const unsigned short* pw0 = ws + OFF_W0;
  const unsigned short* pw1 = ws + OFF_W1;
  const unsigned short* pw2 = ws + OFF_W2;
  const unsigned short* pt0 = ws + OFF_T0;
  const unsigned short* pt1 = ws + OFF_T1;
  const unsigned short* pbd = ws + OFF_BD;
  const unsigned short* psd = ws + OFF_SD;

  // ---- Phase 0: dense input [32,13] -> ACT0 [32,32] bf16 (zero pad) ----
  for (int e = tid; e < 32 * 32; e += 128) {
    int   m = e >> 5, k = e & 31;
    float v = (k < 13) ? dense_x[(size_t)(r0 + m) * 13 + k] : 0.f;
    sACT0[m * 32 + k] = f2bf(v);
  }
  __syncthreads();

  // ---- Bottom MLP ----
  for (int nt = wave; nt < 32; nt += 4)     // 13(pad32) -> 512, relu
    wmma_tile2<true, true>(sACT0, 32, pw0, 32, 32, nt * 16, bb0,
                           sACT1, 512, lane);
  __syncthreads();
  for (int nt = wave; nt < 16; nt += 4)     // 512 -> 256, relu
    wmma_tile2<true, true>(sACT1, 512, pw1, 512, 512, nt * 16, bb1,
                           sACT0, 256, lane);
  __syncthreads();                          // ACT1 dead -> Z region live
  // 256 -> 128 -> Z feature 0 (relu); also zero Z pad features 9..15
  for (int nt = wave; nt < 8; nt += 4)
    wmma_tile2<true, true>(sACT0, 256, pw2, 256, 256, nt * 16, bb2,
                           sZ + 0 * 128, ZS, lane);
  {
    const u32x4 z4 = {0u, 0u, 0u, 0u};
    for (int e = tid; e < 32 * 7 * 16; e += 128) {  // rows x padfeat x 16B chunks
      int m = e / 112, r = e % 112, f = 9 + r / 16, q = r % 16;
      *(u32x4*)(sZ + (size_t)m * ZS + f * 128 + q * 8) = z4;
    }
  }
  __syncthreads();

  // ---- Big AE tables: gather+pool [32,32] then decode @ [32->128] ----
  for (int t = 0; t < NBIG; ++t) {
    const float* tbl = big_tables + (size_t)t * 1000000 * DCMP;
    const int*   ix  = idx_big + ((size_t)t * B + r0) * L;
    for (int e = tid; e < 32 * 8; e += 128) {       // 32 rows x 8 quad-chunks
      int m = e >> 3, d0 = (e & 7) * 4;
      f32x4 s = {0.f, 0.f, 0.f, 0.f};
#pragma unroll
      for (int l = 0; l < L; ++l)
        s += *(const f32x4*)(tbl + (size_t)ix[m * L + l] * DCMP + d0);
#pragma unroll
      for (int q = 0; q < 4; ++q) sACT0[m * DCMP + d0 + q] = f2bf(s[q]);
    }
    __syncthreads();
    for (int nt = wave; nt < 8; nt += 4)
      wmma_tile2<false, false>(sACT0, DCMP, pbd + (size_t)t * 128 * DCMP, DCMP,
                               DCMP, nt * 16, nullptr,
                               sZ + (1 + t) * 128, ZS, lane);
    __syncthreads();
  }

  // ---- Small tables: gather+pool [32,128] then decode @ [128->128] ----
  for (int t = 0; t < NSML; ++t) {
    const float* tbl = small_tables + (size_t)t * 5000 * D;
    const int*   ix  = idx_small + ((size_t)t * B + r0) * L;
    for (int e = tid; e < 32 * 32; e += 128) {      // 32 rows x 32 quad-chunks
      int m = e >> 5, d0 = (e & 31) * 4;
      f32x4 s = {0.f, 0.f, 0.f, 0.f};
#pragma unroll
      for (int l = 0; l < L; ++l)
        s += *(const f32x4*)(tbl + (size_t)ix[m * L + l] * D + d0);
#pragma unroll
      for (int q = 0; q < 4; ++q) sACT0[m * D + d0 + q] = f2bf(s[q]);
    }
    __syncthreads();
    for (int nt = wave; nt < 8; nt += 4)
      wmma_tile2<false, false>(sACT0, D, psd + (size_t)t * D * D, D,
                               D, nt * 16, nullptr,
                               sZ + (5 + t) * 128, ZS, lane);
    __syncthreads();
  }

  // ---- Build R (aliases ACT0): copy x + zero pad, then WMMA interaction ----
  for (int e = tid; e < 32 * 24; e += 128) {        // 24 x 16B chunks per row
    int m = e / 24, cq = e % 24;
    u32x4 v = {0u, 0u, 0u, 0u};
    if (cq < 16) v = *(const u32x4*)(sZ + (size_t)m * ZS + cq * 8);
    *(u32x4*)(sR + (size_t)m * 192 + cq * 8) = v;
  }
  __syncthreads();
  for (int m = wave; m < 32; m += 4)                // S = Z Z^T -> R[128..163]
    interact_row(sZ + (size_t)m * ZS, sR + (size_t)m * 192 + 128, lane);
  __syncthreads();

  // ---- Top MLP ----
  for (int nt = wave; nt < 32; nt += 4)     // 164(pad192) -> 512, relu (Z dead)
    wmma_tile2<true, true>(sR, 192, pt0, 192, 192, nt * 16, tb0,
                           sACT1, 512, lane);
  __syncthreads();
  for (int nt = wave; nt < 16; nt += 4)     // 512 -> 256, relu (R dead)
    wmma_tile2<true, true>(sACT1, 512, pt1, 512, 512, nt * 16, tb1,
                           sACT0, 256, lane);
  __syncthreads();

  // ---- Final 256 -> 1 dot + sigmoid ----
  if (tid < 32) {
    float z = tb2[0];
    for (int d = 0; d < 256; ++d) z += bf2f(sACT0[tid * 256 + d]) * tw2[d];
    out[r0 + tid] = 1.f / (1.f + __expf(-z));
  }
}

// ---------------------------------------------------------------------------

extern "C" void kernel_launch(void* const* d_in, const int* in_sizes, int n_in,
                              void* d_out, int out_size, void* d_ws, size_t ws_size,
                              hipStream_t stream) {
  (void)in_sizes; (void)n_in; (void)out_size; (void)ws_size;
  const float* dense_x      = (const float*)d_in[0];
  const int*   idx_big      = (const int*)  d_in[1];
  const int*   idx_small    = (const int*)  d_in[2];
  const float* big_tables   = (const float*)d_in[3];
  const float* big_dec      = (const float*)d_in[4];
  const float* small_tables = (const float*)d_in[5];
  const float* small_dec    = (const float*)d_in[6];
  const float* bw0 = (const float*)d_in[7];
  const float* bb0 = (const float*)d_in[8];
  const float* bw1 = (const float*)d_in[9];
  const float* bb1 = (const float*)d_in[10];
  const float* bw2 = (const float*)d_in[11];
  const float* bb2 = (const float*)d_in[12];
  const float* tw0 = (const float*)d_in[13];
  const float* tb0 = (const float*)d_in[14];
  const float* tw1 = (const float*)d_in[15];
  const float* tb1 = (const float*)d_in[16];
  const float* tw2 = (const float*)d_in[17];
  const float* tb2 = (const float*)d_in[18];
  float*          out = (float*)d_out;
  unsigned short* ws  = (unsigned short*)d_ws;   // 983,040 bytes used

  // Kernel 1: repack weights fp32 -> padded/transposed bf16 in d_ws.
  repack_weights<<<512, 256, 0, stream>>>(bw0, bw1, bw2, tw0, tw1,
                                          big_dec, small_dec, ws);
  // Kernel 2: fused forward. 4096 rows / 32 per block -> 128 blocks.
  dlrm_fused<<<128, 128, 0, stream>>>(dense_x, idx_big, idx_small,
                                      big_tables, small_tables, ws,
                                      bb0, bb1, bb2, tb0, tb1, tw2, tb2, out);
}
